// CausalSelfAttention_72232759984695
// MI455X (gfx1250) — matrix operand order repaired and checked by reading, not compiled, not run on hardware
//
#include <hip/hip_runtime.h>
#include <hip/hip_bf16.h>

// ---------------------------------------------------------------------------
// CausalSelfAttention forward for MI455X (gfx1250), wave32 + WMMA bf16.
// B=4, T=2048, C=1024, H=16, D=64.
// ---------------------------------------------------------------------------

#define TT   2048
#define CC   1024
#define HH   16
#define DD   64
#define N3   3072
#define MM   8192   // B*T

typedef __bf16 bf16_t;
typedef __attribute__((ext_vector_type(16))) __bf16 bf16x16;
typedef __attribute__((ext_vector_type(8)))  __bf16 bf16x8;
typedef __attribute__((ext_vector_type(8)))  float  f32x8;

union FragU { bf16x16 v; bf16x8 h[2]; };

__device__ inline bf16x16 join8(bf16x8 lo, bf16x8 hi) {
    FragU u; u.h[0] = lo; u.h[1] = hi; return u.v;
}

__device__ inline f32x8 zero8() {
    f32x8 z = {0.f, 0.f, 0.f, 0.f, 0.f, 0.f, 0.f, 0.f};
    return z;
}

__device__ inline f32x8 wmma_bf16(bf16x16 a, bf16x16 b, f32x8 c) {
    // (neg_a, A, neg_b, B, c_mod, C, reuse_a, reuse_b)
    return __builtin_amdgcn_wmma_f32_16x16x32_bf16(false, a, false, b,
                                                   (short)0, c, false, false);
}

// ---------------------------------------------------------------------------
// Kernel 1: qkv = x @ W_qkv + b_qkv, scattered into q[B,H,T,D], k[B,H,T,D],
// v^T [B,H,D,T], all bf16. Block tile 128(M) x 128(N), 8 waves (4x2),
// each wave computes 32x64 = 2x4 WMMA tiles. Double-buffered LDS pipeline,
// one barrier per K-step of 32.
// ---------------------------------------------------------------------------
__global__ __launch_bounds__(256) void qkv_proj_kernel(
    const float* __restrict__ x, const float* __restrict__ Wqkv,
    const float* __restrict__ bqkv,
    bf16_t* __restrict__ qb, bf16_t* __restrict__ kb, bf16_t* __restrict__ vtb)
{
    __shared__ __align__(16) bf16_t Abuf[2][128][32];   // [m][k]
    __shared__ __align__(16) bf16_t Bbuf[2][128][32];   // [n][k] (transposed)

    const int tid  = threadIdx.x;
    const int wave = tid >> 5, lane = tid & 31;
    const int wm = wave >> 1, wn = wave & 1;            // 4 x 2 wave grid
    const int hi = lane >> 4, l16 = lane & 15;
    const int bm = blockIdx.y * 128;
    const int bn = blockIdx.x * 128;

    // staging addresses (constant across K-steps)
    const int arow = tid >> 3, ac4 = (tid & 7) * 4;     // + i*32 rows
    const int bkr  = tid >> 6, bn4 = (tid & 63) * 2;    // W: 32 rows x 128 cols
    // (re-derive per-i below to keep 4 chunks per thread)

    f32x8 acc[2][4];
    #pragma unroll
    for (int mi = 0; mi < 2; ++mi)
        #pragma unroll
        for (int ni = 0; ni < 4; ++ni) acc[mi][ni] = zero8();

    float4 areg[4], breg[4];

    auto load_global = [&](int kk) {
        #pragma unroll
        for (int i = 0; i < 4; ++i) {
            int idx = tid + i * 256;
            int row = idx >> 3, c4 = (idx & 7) * 4;
            areg[i] = *(const float4*)(x + (size_t)(bm + row) * CC + kk + c4);
        }
        #pragma unroll
        for (int i = 0; i < 4; ++i) {
            int idx = tid + i * 256;
            int kr = idx >> 5, n4 = (idx & 31) * 4;
            breg[i] = *(const float4*)(Wqkv + (size_t)(kk + kr) * N3 + bn + n4);
        }
    };
    auto store_lds = [&](int s) {
        #pragma unroll
        for (int i = 0; i < 4; ++i) {
            int idx = tid + i * 256;
            int row = idx >> 3, c4 = (idx & 7) * 4;
            Abuf[s][row][c4 + 0] = (bf16_t)areg[i].x;
            Abuf[s][row][c4 + 1] = (bf16_t)areg[i].y;
            Abuf[s][row][c4 + 2] = (bf16_t)areg[i].z;
            Abuf[s][row][c4 + 3] = (bf16_t)areg[i].w;
        }
        #pragma unroll
        for (int i = 0; i < 4; ++i) {
            int idx = tid + i * 256;
            int kr = idx >> 5, n4 = (idx & 31) * 4;
            Bbuf[s][n4 + 0][kr] = (bf16_t)breg[i].x;
            Bbuf[s][n4 + 1][kr] = (bf16_t)breg[i].y;
            Bbuf[s][n4 + 2][kr] = (bf16_t)breg[i].z;
            Bbuf[s][n4 + 3][kr] = (bf16_t)breg[i].w;
        }
    };

    load_global(0);
    store_lds(0);
    __syncthreads();

    const int NIT = CC / 32;
    for (int it = 0; it < NIT; ++it) {
        const int cur = it & 1;
        if (it + 1 < NIT) load_global((it + 1) * 32);

        bf16x16 afrag[2], bfrag[4];
        #pragma unroll
        for (int mi = 0; mi < 2; ++mi) {
            int r  = wm * 32 + mi * 16 + l16;
            int k0 = hi * 8;                            // A: chunks k0, k0+16
            afrag[mi] = join8(*(const bf16x8*)&Abuf[cur][r][k0],
                              *(const bf16x8*)&Abuf[cur][r][k0 + 16]);
        }
        #pragma unroll
        for (int ni = 0; ni < 4; ++ni) {
            int n  = wn * 64 + ni * 16 + l16;
            int k0 = hi * 8;                            // B: chunks k0, k0+16
            bfrag[ni] = join8(*(const bf16x8*)&Bbuf[cur][n][k0],
                              *(const bf16x8*)&Bbuf[cur][n][k0 + 16]);
        }
        #pragma unroll
        for (int mi = 0; mi < 2; ++mi)
            #pragma unroll
            for (int ni = 0; ni < 4; ++ni)
                acc[mi][ni] = wmma_bf16(afrag[mi], bfrag[ni], acc[mi][ni]);

        if (it + 1 < NIT) store_lds(1 - cur);
        __syncthreads();
    }
    (void)arow; (void)ac4; (void)bkr; (void)bn4;

    // epilogue: block tile lies wholly in one of q/k/v (128 | 1024)
    const int seg = bn / CC;                            // 0=q, 1=k, 2=v (uniform)
    bf16_t* __restrict__ dst = (seg == 0) ? qb : ((seg == 1) ? kb : vtb);
    #pragma unroll
    for (int ni = 0; ni < 4; ++ni) {
        const int n = bn + wn * 64 + ni * 16 + l16;
        const int c = n & (CC - 1);
        const int h = c >> 6, d = c & 63;
        const float bias = bqkv[n];
        #pragma unroll
        for (int mi = 0; mi < 2; ++mi) {
            #pragma unroll
            for (int r = 0; r < 8; ++r) {
                int m = bm + wm * 32 + mi * 16 + r + hi * 8;
                int b = m >> 11, t = m & (TT - 1);
                bf16_t o = (bf16_t)(acc[mi][ni][r] + bias);
                if (seg < 2) {
                    dst[(((size_t)b * HH + h) * TT + t) * DD + d] = o;      // q,k
                } else {
                    dst[(((size_t)b * HH + h) * DD + d) * TT + t] = o;      // V^T
                }
            }
        }
    }
}

// ---------------------------------------------------------------------------
// Kernel 2: flash attention. One WG = (b,h, 64-row q block), 4 waves,
// wave w owns a 16-row q tile. KV loop in blocks of 64 (causal).
// ---------------------------------------------------------------------------
__global__ __launch_bounds__(128) void attn_kernel(
    const bf16_t* __restrict__ qg, const bf16_t* __restrict__ kg,
    const bf16_t* __restrict__ vtg, bf16_t* __restrict__ aog)
{
    __shared__ __align__(16) float  Sbuf[4][16][68];
    __shared__ __align__(16) bf16_t Pbuf[4][16][64];
    __shared__ float Arow[4][16];

    const int wave = threadIdx.x >> 5, lane = threadIdx.x & 31;
    const int hi = lane >> 4, l16 = lane & 15;
    const int bh   = blockIdx.x >> 5;        // T/64 = 32 q blocks
    const int qblk = blockIdx.x & 31;
    const int b = bh >> 4, h = bh & 15;
    const int qrow0 = qblk * 64 + wave * 16;

    const bf16_t* qp = qg  + (size_t)bh * TT * DD;   // [T][D]
    const bf16_t* kp = kg  + (size_t)bh * TT * DD;   // [T][D]
    const bf16_t* vp = vtg + (size_t)bh * DD * TT;   // [D][T]

    // preload q A-fragments (two K=32 chunks over D=64)
    bf16x16 qf[2];
    {
        int row = qrow0 + l16;
        #pragma unroll
        for (int dc = 0; dc < 2; ++dc) {
            int d0 = dc * 32 + hi * 8;
            qf[dc] = join8(*(const bf16x8*)(qp + (size_t)row * DD + d0),
                           *(const bf16x8*)(qp + (size_t)row * DD + d0 + 16));
        }
    }

    f32x8 of[4];
    #pragma unroll
    for (int j = 0; j < 4; ++j) of[j] = zero8();
    float m_run = -INFINITY, l_run = 0.f;    // meaningful on lanes 0..15

    for (int kb = 0; kb <= qblk; ++kb) {
        const int kv0 = kb * 64;
        // ---- S = (q . k^T) / sqrt(D), 4 tiles of 16 kv columns ----
        #pragma unroll
        for (int j = 0; j < 4; ++j) {
            f32x8 s = zero8();
            #pragma unroll
            for (int dc = 0; dc < 2; ++dc) {
                int kvcol = kv0 + j * 16 + l16;
                int d0 = dc * 32 + hi * 8;           // B frag: chunks d0, d0+16
                bf16x16 bf = join8(*(const bf16x8*)(kp + (size_t)kvcol * DD + d0),
                                   *(const bf16x8*)(kp + (size_t)kvcol * DD + d0 + 16));
                s = wmma_bf16(qf[dc], bf, s);
            }
            #pragma unroll
            for (int r = 0; r < 8; ++r) {
                int ml  = r + hi * 8;
                int qr  = qrow0 + ml;
                int kvg = kv0 + j * 16 + l16;
                float sv = s[r] * 0.125f;            // 1/sqrt(64)
                if (kvg > qr) sv = -INFINITY;        // causal mask
                Sbuf[wave][ml][j * 16 + l16] = sv;
            }
        }
        __syncthreads();
        // ---- online softmax (lanes 0..15, one row each) ----
        if (lane < 16) {
            int row = l16;
            float mmax = m_run;
            #pragma unroll 8
            for (int c = 0; c < 64; ++c) mmax = fmaxf(mmax, Sbuf[wave][row][c]);
            float alpha = __expf(m_run - mmax);
            float sum = 0.f;
            #pragma unroll 8
            for (int c = 0; c < 64; ++c) {
                float p = __expf(Sbuf[wave][row][c] - mmax);
                sum += p;
                Pbuf[wave][row][c] = (bf16_t)p;
            }
            l_run = l_run * alpha + sum;
            m_run = mmax;
            Arow[wave][row] = alpha;
        }
        __syncthreads();
        // ---- rescale O accumulators ----
        #pragma unroll
        for (int r = 0; r < 8; ++r) {
            float a = Arow[wave][r + hi * 8];
            #pragma unroll
            for (int j = 0; j < 4; ++j) of[j][r] *= a;
        }
        // ---- O += P @ V  (A frags from LDS, B frags from V^T in L2) ----
        #pragma unroll
        for (int kc = 0; kc < 2; ++kc) {
            int k0 = kc * 32 + hi * 8;
            bf16x16 pf = join8(*(const bf16x8*)&Pbuf[wave][l16][k0],
                               *(const bf16x8*)&Pbuf[wave][l16][k0 + 16]);
            #pragma unroll
            for (int j = 0; j < 4; ++j) {
                int d   = j * 16 + l16;
                int kvs = kv0 + kc * 32 + hi * 8;
                bf16x16 vf = join8(*(const bf16x8*)(vp + (size_t)d * TT + kvs),
                                   *(const bf16x8*)(vp + (size_t)d * TT + kvs + 16));
                of[j] = wmma_bf16(pf, vf, of[j]);
            }
        }
        __syncthreads();
    }

    // ---- finalize: O /= l, store bf16 into [B,T,C] for out-proj ----
    if (lane < 16) Arow[wave][l16] = 1.0f / l_run;
    __syncthreads();
    #pragma unroll
    for (int r = 0; r < 8; ++r) {
        int ml = r + hi * 8;
        float linv = Arow[wave][ml];
        int t = qrow0 + ml;
        #pragma unroll
        for (int j = 0; j < 4; ++j) {
            int d = j * 16 + l16;
            aog[((size_t)(b * TT + t)) * CC + h * DD + d] = (bf16_t)(of[j][r] * linv);
        }
    }
}

// ---------------------------------------------------------------------------
// Kernel 3: out = attn_out(bf16) @ W_out + b_out -> fp32 [B,T,C].
// Same double-buffered pipeline as kernel 1.
// ---------------------------------------------------------------------------
__global__ __launch_bounds__(256) void out_proj_kernel(
    const bf16_t* __restrict__ ao, const float* __restrict__ Wout,
    const float* __restrict__ bout, float* __restrict__ out)
{
    __shared__ __align__(16) bf16_t Abuf[2][128][32];   // [m][k]
    __shared__ __align__(16) bf16_t Bbuf[2][128][32];   // [n][k]

    const int tid  = threadIdx.x;
    const int wave = tid >> 5, lane = tid & 31;
    const int wm = wave >> 1, wn = wave & 1;
    const int hi = lane >> 4, l16 = lane & 15;
    const int bm = blockIdx.y * 128;
    const int bn = blockIdx.x * 128;

    f32x8 acc[2][4];
    #pragma unroll
    for (int mi = 0; mi < 2; ++mi)
        #pragma unroll
        for (int ni = 0; ni < 4; ++ni) acc[mi][ni] = zero8();

    uint4 areg[2];
    float4 breg[4];

    auto load_global = [&](int kk) {
        #pragma unroll
        for (int i = 0; i < 2; ++i) {
            int idx = tid + i * 256;
            int row = idx >> 2, c8 = (idx & 3) * 8;
            areg[i] = *(const uint4*)(ao + (size_t)(bm + row) * CC + kk + c8);
        }
        #pragma unroll
        for (int i = 0; i < 4; ++i) {
            int idx = tid + i * 256;
            int kr = idx >> 5, n4 = (idx & 31) * 4;
            breg[i] = *(const float4*)(Wout + (size_t)(kk + kr) * CC + bn + n4);
        }
    };
    auto store_lds = [&](int s) {
        #pragma unroll
        for (int i = 0; i < 2; ++i) {
            int idx = tid + i * 256;
            int row = idx >> 2, c8 = (idx & 3) * 8;
            *(uint4*)&Abuf[s][row][c8] = areg[i];
        }
        #pragma unroll
        for (int i = 0; i < 4; ++i) {
            int idx = tid + i * 256;
            int kr = idx >> 5, n4 = (idx & 31) * 4;
            Bbuf[s][n4 + 0][kr] = (bf16_t)breg[i].x;
            Bbuf[s][n4 + 1][kr] = (bf16_t)breg[i].y;
            Bbuf[s][n4 + 2][kr] = (bf16_t)breg[i].z;
            Bbuf[s][n4 + 3][kr] = (bf16_t)breg[i].w;
        }
    };

    load_global(0);
    store_lds(0);
    __syncthreads();

    const int NIT = CC / 32;
    for (int it = 0; it < NIT; ++it) {
        const int cur = it & 1;
        if (it + 1 < NIT) load_global((it + 1) * 32);

        bf16x16 afrag[2], bfrag[4];
        #pragma unroll
        for (int mi = 0; mi < 2; ++mi) {
            int r  = wm * 32 + mi * 16 + l16;
            int k0 = hi * 8;
            afrag[mi] = join8(*(const bf16x8*)&Abuf[cur][r][k0],
                              *(const bf16x8*)&Abuf[cur][r][k0 + 16]);
        }
        #pragma unroll
        for (int ni = 0; ni < 4; ++ni) {
            int n  = wn * 64 + ni * 16 + l16;
            int k0 = hi * 8;
            bfrag[ni] = join8(*(const bf16x8*)&Bbuf[cur][n][k0],
                              *(const bf16x8*)&Bbuf[cur][n][k0 + 16]);
        }
        #pragma unroll
        for (int mi = 0; mi < 2; ++mi)
            #pragma unroll
            for (int ni = 0; ni < 4; ++ni)
                acc[mi][ni] = wmma_bf16(afrag[mi], bfrag[ni], acc[mi][ni]);

        if (it + 1 < NIT) store_lds(1 - cur);
        __syncthreads();
    }

    #pragma unroll
    for (int ni = 0; ni < 4; ++ni) {
        const int n = bn + wn * 64 + ni * 16 + l16;
        const float bias = bout[n];
        #pragma unroll
        for (int mi = 0; mi < 2; ++mi) {
            #pragma unroll
            for (int r = 0; r < 8; ++r) {
                int m = bm + wm * 32 + mi * 16 + r + hi * 8;
                out[(size_t)m * CC + n] = acc[mi][ni][r] + bias;
            }
        }
    }
}

// ---------------------------------------------------------------------------
extern "C" void kernel_launch(void* const* d_in, const int* in_sizes, int n_in,
                              void* d_out, int out_size, void* d_ws, size_t ws_size,
                              hipStream_t stream) {
    (void)in_sizes; (void)n_in; (void)out_size; (void)ws_size;
    const float* x    = (const float*)d_in[0];
    const float* Wqkv = (const float*)d_in[1];
    const float* bqkv = (const float*)d_in[2];
    const float* Wout = (const float*)d_in[3];
    const float* bout = (const float*)d_in[4];
    float* out = (float*)d_out;

    const size_t seg = (size_t)4 * HH * TT * DD;     // 8M bf16 elements
    bf16_t* qb  = (bf16_t*)d_ws;
    bf16_t* kb  = qb  + seg;
    bf16_t* vtb = kb  + seg;
    bf16_t* aob = vtb + seg;                          // [B*T, C] bf16

    qkv_proj_kernel<<<dim3(N3 / 128, MM / 128), 256, 0, stream>>>(
        x, Wqkv, bqkv, qb, kb, vtb);
    attn_kernel<<<dim3(4 * HH * (TT / 64)), 128, 0, stream>>>(
        qb, kb, vtb, aob);
    out_proj_kernel<<<dim3(CC / 128, MM / 128), 256, 0, stream>>>(
        aob, Wout, bout, out);
}
